// DecoderLayer_69672959475717
// MI455X (gfx1250) — matrix operand order repaired and checked by reading, not compiled, not run on hardware
//
#include <hip/hip_runtime.h>
#include <hip/hip_bf16.h>

// ---------------- problem constants ----------------
#define BATCH 2
#define SEQ   2048
#define DIM   1024
#define NH    16
#define HDIM  64
#define NEXP  8
#define TOPK  2
#define HID   4096
#define BS    (BATCH * SEQ)   // 4096 token rows

typedef __attribute__((ext_vector_type(16))) __bf16 v16bf;
typedef __attribute__((ext_vector_type(8)))  __bf16 v8bf;
typedef __attribute__((ext_vector_type(8)))  float  v8f;
typedef __attribute__((ext_vector_type(4))) unsigned int v4u;
typedef __attribute__((ext_vector_type(8))) int          v8i;
typedef __attribute__((ext_vector_type(4))) int          v4i;

#define AS1 __attribute__((address_space(1)))
#define AS3 __attribute__((address_space(3)))

// ---- gfx1250 data-mover feature probes (compile-safe on any toolchain) ----
#if __has_builtin(__builtin_amdgcn_tensor_load_to_lds)
#define HAVE_TDM 1
#else
#define HAVE_TDM 0
#endif
#if __has_builtin(__builtin_amdgcn_global_load_async_to_lds_b128)
#define HAVE_ASYNC 1
#else
#define HAVE_ASYNC 0
#endif

#if HAVE_TDM
// Build a D# for a 2D bf16 tile (tile_d0 contiguous elems x tile_d1 rows,
// row stride stride0 elems) and issue TENSOR_LOAD_TO_LDS (TDM DMA).
__device__ inline void tdm_load_tile_2d_bf16(const __bf16* gsrc, unsigned lds_byte_off,
                                             unsigned tensor_d0, unsigned tensor_d1,
                                             unsigned tile_d0, unsigned tile_d1,
                                             unsigned long long stride0) {
  unsigned long long ga = (unsigned long long)(const void*)gsrc;
  v4u g0;
  g0[0] = 1u;                                              // count=1, user descriptor
  g0[1] = lds_byte_off;                                    // lds_addr (bytes)
  g0[2] = (unsigned)ga;                                    // global_addr[31:0]
  g0[3] = (unsigned)((ga >> 32) & 0x01ffffffu) | (2u << 30); // addr[56:32] | type=2
  v8i g1;
  g1[0] = (int)(1u << 16);                                 // wg_mask=0, data_size=1 (2B)
  g1[1] = (int)((tensor_d0 & 0xffffu) << 16);              // tensor_dim0[15:0] @ 63:48
  g1[2] = (int)((tensor_d0 >> 16) | ((tensor_d1 & 0xffffu) << 16));
  g1[3] = (int)((tensor_d1 >> 16) | ((tile_d0 & 0xffffu) << 16)); // tile_dim0 @127:112
  g1[4] = (int)(tile_d1 & 0xffffu);                        // tile_dim1, tile_dim2=0
  g1[5] = (int)(unsigned)(stride0 & 0xffffffffu);          // tensor_dim0_stride[31:0]
  g1[6] = (int)(unsigned)((stride0 >> 32) & 0xffffu);      // stride[47:32], dim1_stride=0
  g1[7] = 0;
  v4i z4 = {0, 0, 0, 0};                                   // groups 2/3 unused (2D)
  v8i z8 = {0, 0, 0, 0, 0, 0, 0, 0};
  __builtin_amdgcn_tensor_load_to_lds(g0, g1, z4, z4, z8, 0);
}

__device__ inline void tdm_wait0() {
#if __has_builtin(__builtin_amdgcn_s_wait_tensorcnt)
  __builtin_amdgcn_s_wait_tensorcnt(0);
#else
  asm volatile("s_wait_tensorcnt 0" ::: "memory");
#endif
}
#endif  // HAVE_TDM

#if HAVE_ASYNC
__device__ inline void async_copy_b128(const __bf16* gsrc, __bf16* lds_dst) {
  __builtin_amdgcn_global_load_async_to_lds_b128(
      (AS1 v4i*)(const void*)gsrc, (AS3 v4i*)(void*)lds_dst, 0, 0);
}
__device__ inline void async_wait0() {
#if __has_builtin(__builtin_amdgcn_s_wait_asynccnt)
  __builtin_amdgcn_s_wait_asynccnt(0);
#else
  asm volatile("s_wait_asynccnt 0" ::: "memory");
#endif
}
#endif  // HAVE_ASYNC

// ---------------- helpers ----------------
__device__ inline v8f wmma_bf16(v16bf a, v16bf b, v8f c) {
  // D = A(16x32 bf16) * B(32x16 bf16) + C(16x16 f32)
  return __builtin_amdgcn_wmma_f32_16x16x32_bf16(
      /*neg_a=*/false, a, /*neg_b=*/false, b,
      /*c_mod=*/(short)0, c, /*reuse_a=*/false, /*reuse_b=*/false);
}

__device__ inline v16bf ldfrag(const __bf16* p0, const __bf16* p1) {
  v8bf a = *(const v8bf*)p0;
  v8bf b = *(const v8bf*)p1;
  v16bf r;
#pragma unroll
  for (int i = 0; i < 8; ++i) { r[i] = a[i]; r[8 + i] = b[i]; }
  return r;
}

__device__ inline float gelu_tanh(float x) {
  const float c0 = 0.7978845608028654f;   // sqrt(2/pi)
  float x3 = x * x * x;
  return 0.5f * x * (1.0f + tanhf(c0 * (x + 0.044715f * x3)));
}

// ---------------- elementwise / norm kernels ----------------
__global__ __launch_bounds__(256)
void cvt_f32_to_bf16_kernel(const float* __restrict__ in, __bf16* __restrict__ out, size_t n) {
  size_t i = (size_t)blockIdx.x * 256 + threadIdx.x;
  size_t stride = (size_t)gridDim.x * 256;
  for (; i < n; i += stride) out[i] = (__bf16)in[i];
}

// out_bf16[row] = rmsnorm(x[row]) * g   (row length DIM)
__global__ __launch_bounds__(256)
void rmsnorm_bf16_kernel(const float* __restrict__ x, const float* __restrict__ g,
                         __bf16* __restrict__ out) {
  __shared__ float red[8];
  __shared__ float scale_s;
  const int row = blockIdx.x, tid = threadIdx.x;
  const float* xr = x + (size_t)row * DIM;
  float ss = 0.f;
  for (int i = tid; i < DIM; i += 256) { float v = xr[i]; ss += v * v; }
#pragma unroll
  for (int m = 16; m >= 1; m >>= 1) ss += __shfl_xor(ss, m, 32);
  if ((tid & 31) == 0) red[tid >> 5] = ss;
  __syncthreads();
  if (tid == 0) {
    float s = 0.f;
#pragma unroll
    for (int i = 0; i < 8; ++i) s += red[i];
    scale_s = rsqrtf(s / (float)DIM + 1e-5f);
  }
  __syncthreads();
  const float sc = scale_s;
  for (int i = tid; i < DIM; i += 256)
    out[(size_t)row * DIM + i] = (__bf16)(xr[i] * sc * g[i]);
}

// out_f32[row] = xin[row] + rmsnorm(t[row]) * g
__global__ __launch_bounds__(256)
void add_rmsnorm_kernel(const float* __restrict__ xin, const float* __restrict__ t,
                        const float* __restrict__ g, float* __restrict__ out) {
  __shared__ float red[8];
  __shared__ float scale_s;
  const int row = blockIdx.x, tid = threadIdx.x;
  const float* tr = t + (size_t)row * DIM;
  const float* xr = xin + (size_t)row * DIM;
  float ss = 0.f;
  for (int i = tid; i < DIM; i += 256) { float v = tr[i]; ss += v * v; }
#pragma unroll
  for (int m = 16; m >= 1; m >>= 1) ss += __shfl_xor(ss, m, 32);
  if ((tid & 31) == 0) red[tid >> 5] = ss;
  __syncthreads();
  if (tid == 0) {
    float s = 0.f;
#pragma unroll
    for (int i = 0; i < 8; ++i) s += red[i];
    scale_s = rsqrtf(s / (float)DIM + 1e-5f);
  }
  __syncthreads();
  const float sc = scale_s;
  for (int i = tid; i < DIM; i += 256)
    out[(size_t)row * DIM + i] = xr[i] + tr[i] * sc * g[i];
}

// ---------------- tiled WMMA GEMM ----------------
// C[M,N] = A[M,K](bf16, row major) * B[K,N](bf16, row major)
// block = 128 threads (4 waves); tile 64x64, K-step 32.
// A tile staged by the Tensor Data Mover when available; B tile staged
// transposed so each lane's B fragment is two contiguous 8-value K runs at 8*hsel and 16+8*hsel.
// MODE 0: store f32    MODE 1: store bf16    MODE 2: gelu -> bf16 (block-skip)
// MODE 3: atomic f32 accumulate scaled by routing[row*NEXP+expert] (block-skip)
template <int MODE>
__global__ __launch_bounds__(128)
void gemm_bf16_kernel(const __bf16* __restrict__ A, const __bf16* __restrict__ B,
                      void* __restrict__ Cout, int M, int N, int K,
                      const float* __restrict__ routing, int expert) {
  __shared__ __bf16 As[64][32];       // row stride 64B (16B aligned chunks)
  __shared__ __bf16 Bt[64][40];       // transposed B tile, row stride 80B
  __shared__ int skipflag;

  const int tid  = threadIdx.x;
  const int wave = tid >> 5;
  const int lane = tid & 31;
  const int r16  = lane & 15;
  const int hsel = lane >> 4;          // 0: low half, 1: high half
  const int rowBase = blockIdx.y * 64;
  const int colBase = blockIdx.x * 64;

  if (MODE >= 2) {
    if (tid == 0) skipflag = 0;
    __syncthreads();
    if (tid < 64 && routing[(size_t)(rowBase + tid) * NEXP + expert] > 0.f) skipflag = 1;
    __syncthreads();
    if (!skipflag) return;             // uniform exit: EXEC stays all-1 around WMMA
  }

  v8f acc[4] = {};                      // wave computes 16 rows x 64 cols

  for (int k0 = 0; k0 < K; k0 += 32) {
    // ---- stage A tile 64x32 ----
#if HAVE_TDM
    if (wave == 0) {                    // one TDM issue per block (EXEC-independent DMA)
      tdm_load_tile_2d_bf16(A + (size_t)rowBase * K + k0,
                            (unsigned)(size_t)(&As[0][0]),
                            (unsigned)(K - k0), (unsigned)(M - rowBase),
                            32u, 64u, (unsigned long long)K);
      tdm_wait0();
    }
#else
    {
      const int r = tid >> 1;
      const int c = (tid & 1) * 16;
      const __bf16* src = A + (size_t)(rowBase + r) * K + k0 + c;
      *(v8bf*)&As[r][c]     = *(const v8bf*)src;
      *(v8bf*)&As[r][c + 8] = *(const v8bf*)(src + 8);
    }
#endif
    // ---- stage B tile 32x64 transposed -> Bt[n][k] ----
    {
      const int kr = tid >> 2;
      const int cb = (tid & 3) * 16;
      const __bf16* src = B + (size_t)(k0 + kr) * N + colBase + cb;
      v8bf b0 = *(const v8bf*)src;
      v8bf b1 = *(const v8bf*)(src + 8);
#pragma unroll
      for (int j = 0; j < 8; ++j) {
        Bt[cb + j][kr]     = b0[j];
        Bt[cb + 8 + j][kr] = b1[j];
      }
    }
    __syncthreads();
    if (k0 + 32 < K) {                  // hint next tiles toward L2 (global_prefetch_b8)
      __builtin_prefetch(A + (size_t)(rowBase + (tid >> 1)) * K + k0 + 32, 0, 1);
      __builtin_prefetch(B + (size_t)(k0 + 32 + (tid >> 2)) * N + colBase, 0, 1);
    }
    // A fragment: wave rows [16w, 16w+16); lane half selects K groups per ISA layout
    const int ar = 16 * wave + r16;
    const int ks = hsel * 8;
    v16bf af = ldfrag(&As[ar][ks], &As[ar][16 + ks]);
#pragma unroll
    for (int j = 0; j < 4; ++j) {
      const int br = 16 * j + r16;
      v16bf bf_ = ldfrag(&Bt[br][hsel * 8], &Bt[br][16 + hsel * 8]);
      acc[j] = wmma_bf16(af, bf_, acc[j]);
    }
    __syncthreads();
  }

  // epilogue: C layout -> row = rowBase + 16*wave + i + 8*hsel, col = colBase + 16*j + r16
#pragma unroll
  for (int j = 0; j < 4; ++j) {
#pragma unroll
    for (int i = 0; i < 8; ++i) {
      const int row = rowBase + 16 * wave + i + 8 * hsel;
      const int col = colBase + 16 * j + r16;
      float v = acc[j][i];
      if (MODE == 0) {
        ((float*)Cout)[(size_t)row * N + col] = v;
      } else if (MODE == 1) {
        ((__bf16*)Cout)[(size_t)row * N + col] = (__bf16)v;
      } else if (MODE == 2) {
        ((__bf16*)Cout)[(size_t)row * N + col] = (__bf16)gelu_tanh(v);
      } else {
        float w = routing[(size_t)row * NEXP + expert];
        if (w > 0.f)
          atomicAdd(&((float*)Cout)[(size_t)row * N + col], w * v);
      }
    }
  }
}

// ---------------- flash attention (MQA), one wave per (b,h,16 q rows) --------
__global__ __launch_bounds__(32)
void mqa_flash_kernel(const __bf16* __restrict__ qb,   // [BS, NH*HDIM]
                      const __bf16* __restrict__ kb,   // [BS, HDIM]
                      const __bf16* __restrict__ vb,   // [BS, HDIM]
                      __bf16* __restrict__ ob) {       // [BS, NH*HDIM]
  __shared__ __bf16 Pl[16][40];
  __shared__ __bf16 Vt[64][40];
#if HAVE_ASYNC
  __shared__ __bf16 Vraw[32 * 64];     // raw V tile landed by async DMA
#endif

  const int L = threadIdx.x;
  const int r16 = L & 15, hsel = L >> 4;
  const int qBase = blockIdx.x * 16;
  const int h = blockIdx.y, b = blockIdx.z;

  // Q fragments (rows qBase..+16, hd 0..63), scale 1/sqrt(HDIM) folded in
  const __bf16* qp = qb + (size_t)(b * SEQ + qBase + r16) * (NH * HDIM) + h * HDIM;
  v16bf aq0, aq1;
  {
    const int ks = hsel * 8;
#pragma unroll
    for (int i = 0; i < 8; ++i) {
      aq0[i]     = (__bf16)((float)qp[ks + i] * 0.125f);
      aq0[8 + i] = (__bf16)((float)qp[16 + ks + i] * 0.125f);
      aq1[i]     = (__bf16)((float)qp[32 + ks + i] * 0.125f);
      aq1[8 + i] = (__bf16)((float)qp[48 + ks + i] * 0.125f);
    }
  }

  float m_[8], l_[8];
#pragma unroll
  for (int i = 0; i < 8; ++i) { m_[i] = -1e30f; l_[i] = 0.f; }
  v8f o_[4] = {};

  for (int kt = 0; kt < SEQ; kt += 32) {
#if HAVE_ASYNC
    // kick the V tile (4 KB) toward LDS early: 8 wave-level async b128 copies
    {
      const __bf16* vg = vb + (size_t)(b * SEQ + kt) * HDIM;
#pragma unroll
      for (int c = 0; c < 8; ++c) {
        const int idx = c * 32 + L;                  // 16-byte chunk id (256 total)
        async_copy_b128(vg + idx * 8, &Vraw[idx * 8]);
      }
    }
#endif
    // scores: two 16x16 tiles (keys kt..+16, kt+16..+32)
    v8f s_[2];
#pragma unroll
    for (int half = 0; half < 2; ++half) {
      const __bf16* kp = kb + (size_t)(b * SEQ + kt + half * 16 + r16) * HDIM;
      v8f s = {};
      s = wmma_bf16(aq0, ldfrag(kp + hsel * 8,      kp + 16 + hsel * 8),      s);
      s = wmma_bf16(aq1, ldfrag(kp + 32 + hsel * 8, kp + 48 + hsel * 8), s);
      s_[half] = s;
    }
    // online softmax; row of C tile lives across the 16-lane half
    float alpha[8];
#pragma unroll
    for (int i = 0; i < 8; ++i) {
      float rm = fmaxf(s_[0][i], s_[1][i]);
      rm = fmaxf(rm, __shfl_xor(rm, 1, 32));
      rm = fmaxf(rm, __shfl_xor(rm, 2, 32));
      rm = fmaxf(rm, __shfl_xor(rm, 4, 32));
      rm = fmaxf(rm, __shfl_xor(rm, 8, 32));
      float mn = fmaxf(m_[i], rm);
      alpha[i] = __expf(m_[i] - mn);
      m_[i] = mn;
      float p0 = __expf(s_[0][i] - mn);
      float p1 = __expf(s_[1][i] - mn);
      float rs = p0 + p1;
      rs += __shfl_xor(rs, 1, 32);
      rs += __shfl_xor(rs, 2, 32);
      rs += __shfl_xor(rs, 4, 32);
      rs += __shfl_xor(rs, 8, 32);
      l_[i] = l_[i] * alpha[i] + rs;
      Pl[i + 8 * hsel][r16]      = (__bf16)p0;
      Pl[i + 8 * hsel][16 + r16] = (__bf16)p1;
    }
#pragma unroll
    for (int j = 0; j < 4; ++j)
#pragma unroll
      for (int i = 0; i < 8; ++i) o_[j][i] *= alpha[i];

    // stage V tile transposed: Vt[hd][key_local]
#if HAVE_ASYNC
    async_wait0();
    __syncthreads();
    {
#pragma unroll
      for (int d0 = 0; d0 < 64; d0 += 8) {
        v8bf vv = *(const v8bf*)&Vraw[L * 64 + d0];
#pragma unroll
        for (int j2 = 0; j2 < 8; ++j2) Vt[d0 + j2][L] = vv[j2];
      }
    }
#else
    {
      const __bf16* vp = vb + (size_t)(b * SEQ + kt + L) * HDIM;
#pragma unroll
      for (int d0 = 0; d0 < 64; d0 += 8) {
        v8bf vv = *(const v8bf*)(vp + d0);
#pragma unroll
        for (int j2 = 0; j2 < 8; ++j2) Vt[d0 + j2][L] = vv[j2];
      }
    }
#endif
    __syncthreads();
    // P fragment (16x32 over keys), then PV into 4 hd chunks
    v16bf pf = ldfrag(&Pl[r16][hsel * 8], &Pl[r16][16 + hsel * 8]);
#pragma unroll
    for (int j = 0; j < 4; ++j) {
      const int vr = 16 * j + r16;
      v16bf vf = ldfrag(&Vt[vr][hsel * 8], &Vt[vr][16 + hsel * 8]);
      o_[j] = wmma_bf16(pf, vf, o_[j]);
    }
    __syncthreads();
  }

#pragma unroll
  for (int j = 0; j < 4; ++j)
#pragma unroll
    for (int i = 0; i < 8; ++i) {
      const int row = qBase + i + 8 * hsel;
      const int col = 16 * j + r16;
      ob[(size_t)(b * SEQ + row) * (NH * HDIM) + h * HDIM + col] =
          (__bf16)(o_[j][i] / l_[i]);
    }
}

// ---------------- router: logits, softmax, top-2 scatter ----------------
__global__ __launch_bounds__(32)
void router_kernel(const __bf16* __restrict__ h2, const float* __restrict__ Wg,
                   float* __restrict__ routing, float* __restrict__ probs_out) {
  const int row = blockIdx.x;
  const int L = threadIdx.x;
  const int e = L & 7, part = L >> 3;   // 4 partial sums per expert
  float acc = 0.f;
  const __bf16* hr = h2 + (size_t)row * DIM;
  for (int i = part; i < DIM; i += 4)
    acc += (float)hr[i] * Wg[(size_t)i * NEXP + e];
  acc += __shfl_xor(acc, 8, 32);
  acc += __shfl_xor(acc, 16, 32);
  float logit = acc;
  float mx = logit;
  mx = fmaxf(mx, __shfl_xor(mx, 1, 32));
  mx = fmaxf(mx, __shfl_xor(mx, 2, 32));
  mx = fmaxf(mx, __shfl_xor(mx, 4, 32));
  float ex = __expf(logit - mx);
  float sum = ex;
  sum += __shfl_xor(sum, 1, 32);
  sum += __shfl_xor(sum, 2, 32);
  sum += __shfl_xor(sum, 4, 32);
  float p = ex / sum;
  // rank within the 8-expert group (ties broken by lower index)
  int rank = 0;
#pragma unroll
  for (int d = 1; d < 8; ++d) {
    float other = __shfl_xor(p, d, 32);
    int oe = e ^ d;
    if (other > p || (other == p && oe < e)) rank++;
  }
  if (L < 8) {
    routing[(size_t)row * NEXP + e]  = (rank < TOPK) ? p : 0.f;
    probs_out[(size_t)row * NEXP + e] = p;
  }
}

// ---------------- host orchestration ----------------
extern "C" void kernel_launch(void* const* d_in, const int* in_sizes, int n_in,
                              void* d_out, int out_size, void* d_ws, size_t ws_size,
                              hipStream_t stream) {
  (void)in_sizes; (void)n_in; (void)out_size; (void)ws_size;
  const float* x          = (const float*)d_in[0];
  const float* g_pre_mqa  = (const float*)d_in[1];
  const float* g_post_mqa = (const float*)d_in[2];
  const float* g_pre_moe  = (const float*)d_in[3];
  const float* g_post_moe = (const float*)d_in[4];
  const float* Wq = (const float*)d_in[5];
  const float* Wk = (const float*)d_in[6];
  const float* Wv = (const float*)d_in[7];
  const float* Wo = (const float*)d_in[8];
  const float* Wg = (const float*)d_in[9];
  const float* We1 = (const float*)d_in[10];
  const float* We2 = (const float*)d_in[11];

  char* wsp = (char*)d_ws;
  auto alloc = [&](size_t bytes) -> char* {
    char* p = wsp;
    wsp += (bytes + 255) & ~(size_t)255;
    return p;
  };
  __bf16* wq_bf  = (__bf16*)alloc((size_t)DIM * NH * HDIM * 2);
  __bf16* wk_bf  = (__bf16*)alloc((size_t)DIM * HDIM * 2);
  __bf16* wv_bf  = (__bf16*)alloc((size_t)DIM * HDIM * 2);
  __bf16* wo_bf  = (__bf16*)alloc((size_t)NH * HDIM * DIM * 2);
  __bf16* we1_bf = (__bf16*)alloc((size_t)NEXP * DIM * HID * 2);
  __bf16* we2_bf = (__bf16*)alloc((size_t)NEXP * HID * DIM * 2);
  __bf16* h1     = (__bf16*)alloc((size_t)BS * DIM * 2);
  __bf16* qb     = (__bf16*)alloc((size_t)BS * NH * HDIM * 2);
  __bf16* kb     = (__bf16*)alloc((size_t)BS * HDIM * 2);
  __bf16* vb     = (__bf16*)alloc((size_t)BS * HDIM * 2);
  __bf16* attnb  = (__bf16*)alloc((size_t)BS * NH * HDIM * 2);
  float*  tmpf   = (float*)alloc((size_t)BS * DIM * 4);
  float*  x2     = (float*)alloc((size_t)BS * DIM * 4);
  __bf16* h2     = (__bf16*)alloc((size_t)BS * DIM * 2);
  float*  routing= (float*)alloc((size_t)BS * NEXP * 4);
  __bf16* mid    = (__bf16*)alloc((size_t)BS * HID * 2);
  float*  moeacc = (float*)alloc((size_t)BS * DIM * 4);

  float* out_x     = (float*)d_out;
  float* out_probs = out_x + (size_t)BS * DIM;

  // --- weight precision: fp32 -> bf16 (WMMA-friendly path) ---
  cvt_f32_to_bf16_kernel<<<2048, 256, 0, stream>>>(Wq,  wq_bf,  (size_t)DIM * NH * HDIM);
  cvt_f32_to_bf16_kernel<<<256,  256, 0, stream>>>(Wk,  wk_bf,  (size_t)DIM * HDIM);
  cvt_f32_to_bf16_kernel<<<256,  256, 0, stream>>>(Wv,  wv_bf,  (size_t)DIM * HDIM);
  cvt_f32_to_bf16_kernel<<<2048, 256, 0, stream>>>(Wo,  wo_bf,  (size_t)NH * HDIM * DIM);
  cvt_f32_to_bf16_kernel<<<4096, 256, 0, stream>>>(We1, we1_bf, (size_t)NEXP * DIM * HID);
  cvt_f32_to_bf16_kernel<<<4096, 256, 0, stream>>>(We2, we2_bf, (size_t)NEXP * HID * DIM);

  // --- attention branch ---
  rmsnorm_bf16_kernel<<<BS, 256, 0, stream>>>(x, g_pre_mqa, h1);
  gemm_bf16_kernel<1><<<dim3((NH * HDIM) / 64, BS / 64), 128, 0, stream>>>(
      h1, wq_bf, qb, BS, NH * HDIM, DIM, nullptr, 0);
  gemm_bf16_kernel<1><<<dim3(HDIM / 64, BS / 64), 128, 0, stream>>>(
      h1, wk_bf, kb, BS, HDIM, DIM, nullptr, 0);
  gemm_bf16_kernel<1><<<dim3(HDIM / 64, BS / 64), 128, 0, stream>>>(
      h1, wv_bf, vb, BS, HDIM, DIM, nullptr, 0);
  mqa_flash_kernel<<<dim3(SEQ / 16, NH, BATCH), 32, 0, stream>>>(qb, kb, vb, attnb);
  gemm_bf16_kernel<0><<<dim3(DIM / 64, BS / 64), 128, 0, stream>>>(
      attnb, wo_bf, tmpf, BS, DIM, NH * HDIM, nullptr, 0);
  add_rmsnorm_kernel<<<BS, 256, 0, stream>>>(x, tmpf, g_post_mqa, x2);

  // --- MoE branch ---
  rmsnorm_bf16_kernel<<<BS, 256, 0, stream>>>(x2, g_pre_moe, h2);
  router_kernel<<<BS, 32, 0, stream>>>(h2, Wg, routing, out_probs);
  (void)hipMemsetAsync(moeacc, 0, (size_t)BS * DIM * 4, stream);
  for (int e = 0; e < NEXP; ++e) {
    gemm_bf16_kernel<2><<<dim3(HID / 64, BS / 64), 128, 0, stream>>>(
        h2, we1_bf + (size_t)e * DIM * HID, mid, BS, HID, DIM, routing, e);
    gemm_bf16_kernel<3><<<dim3(DIM / 64, BS / 64), 128, 0, stream>>>(
        mid, we2_bf + (size_t)e * HID * DIM, moeacc, BS, DIM, HID, routing, e);
  }
  add_rmsnorm_kernel<<<BS, 256, 0, stream>>>(x2, moeacc, g_post_moe, out_x);
}
